// Decoder_29102698398189
// MI455X (gfx1250) — matrix operand order repaired and checked by reading, not compiled, not run on hardware
//
#include <hip/hip_runtime.h>

#define NBATCH 32768
#define PBLK   8
#define H1DIM  128
#define H2DIM  512
#define CDIM   224
#define ZDIM   16

typedef __attribute__((ext_vector_type(16))) _Float16 v16h;
typedef __attribute__((ext_vector_type(8)))  _Float16 v8h;
typedef __attribute__((ext_vector_type(8)))  float    v8f;
typedef __attribute__((ext_vector_type(4)))  float    v4f;

union HFrag { v16h v; v8h p[2]; };

static __device__ __forceinline__ v8f wmma_f16(const HFrag& a, const HFrag& b, v8f c) {
  // 8 args: (neg_a, A, neg_b, B, c_mod, C, reuse_a, reuse_b)
  return __builtin_amdgcn_wmma_f32_16x16x32_f16(false, a.v, false, b.v, (short)0, c, false, false);
}

// ---------------------------------------------------------------------------
// Small utility kernels
// ---------------------------------------------------------------------------
__global__ void cvt_f16_kernel(const float* __restrict__ src, _Float16* __restrict__ dst, int n) {
  int i = blockIdx.x * blockDim.x + threadIdx.x;
  if (i < n) dst[i] = (_Float16)src[i];
}

__global__ void zero_f32_kernel(float* __restrict__ dst, int n) {
  int i = blockIdx.x * blockDim.x + threadIdx.x;
  if (i < n) dst[i] = 0.0f;
}

// a = gamma * rsqrt(var+eps); c = beta - mu*a   (biased variance, per reference)
__global__ void bn_finalize_kernel(const float* __restrict__ sum, const float* __restrict__ sqr,
                                   const float* __restrict__ gamma, const float* __restrict__ beta,
                                   float* __restrict__ aOut, float* __restrict__ cOut,
                                   int n, float invN) {
  int i = blockIdx.x * blockDim.x + threadIdx.x;
  if (i < n) {
    float mu  = sum[i] * invN;
    float var = sqr[i] * invN - mu * mu;
    var = var > 0.0f ? var : 0.0f;
    float a = gamma[i] * rsqrtf(var + 1e-5f);
    aOut[i] = a;
    cOut[i] = beta[i] - mu * a;
  }
}

// ---------------------------------------------------------------------------
// GEMM1: h1[p,n,128] = z[n,16] @ W6[p]^T + b6   (K=16, zero-padded to 32)
// grid (N/128, P), block 256 (8 waves); wave w -> rows w*16..w*16+15
// ---------------------------------------------------------------------------
__global__ void __launch_bounds__(256)
gemm1_kernel(const _Float16* __restrict__ zh, const _Float16* __restrict__ W6h,
             const float* __restrict__ b6,
             _Float16* __restrict__ h1, float* __restrict__ gSum, float* __restrict__ gSqr)
{
  __shared__ _Float16 outT[128 * 136];     // repack tile for coalesced b128 stores
  __shared__ float rSum[H1DIM], rSqr[H1DIM];
  const int tid  = threadIdx.x;
  const int lane = tid & 31, wave = tid >> 5;
  const int l15  = lane & 15, hl = lane >> 4;
  const int rowbase = blockIdx.x * 128;
  const int p = blockIdx.y;

  if (tid < H1DIM) { rSum[tid] = 0.0f; rSqr[tid] = 0.0f; }
  __syncthreads();

  v8h zero8;
#pragma unroll
  for (int i = 0; i < 8; ++i) zero8[i] = (_Float16)0.0f;

  // A fragment: lane l holds row (l&15); j<8 -> k=hl*8+j (valid), j>=8 -> k>=16 (zero pad).
  HFrag af;
  af.p[0] = *(const v8h*)&zh[(size_t)(rowbase + wave * 16 + l15) * ZDIM + hl * 8];
  af.p[1] = zero8;

  v8f acc[8];
#pragma unroll
  for (int nt = 0; nt < 8; ++nt) {
#pragma unroll
    for (int i = 0; i < 8; ++i) acc[nt][i] = 0.0f;
    // B fragment: lane l holds col (l&15); j<8 -> k=hl*8+j (valid), j>=8 -> k>=16 (zero pad).
    // Branchless: address is hl-independent, so load for all lanes and select.
    const v8h* bs = (const v8h*)&W6h[(size_t)(p * H1DIM + nt * 16 + l15) * ZDIM];
    v8h q0 = bs[0], q1 = bs[1];
    HFrag bf;
    bf.p[0] = (hl == 0) ? q0 : q1;
    bf.p[1] = zero8;
    acc[nt] = wmma_f16(af, bf, acc[nt]);
  }

  // epilogue: stats + repack into LDS
#pragma unroll
  for (int nt = 0; nt < 8; ++nt) {
    const int n = nt * 16 + l15;
    const float bb = b6[p * H1DIM + n];
    float ls = 0.0f, lq = 0.0f;
#pragma unroll
    for (int r = 0; r < 8; ++r) {
      float raw = acc[nt][r] + bb;
      outT[(wave * 16 + hl * 8 + r) * 136 + n] = (_Float16)raw;
      ls += raw; lq += raw * raw;
    }
    atomicAdd(&rSum[n], ls);
    atomicAdd(&rSqr[n], lq);
  }
  __syncthreads();
  // cooperative coalesced store: 128 rows x 16 b128-chunks
#pragma unroll
  for (int i = 0; i < 8; ++i) {
    const int c = tid + i * 256;
    const int row = c >> 4, seg = c & 15;
    v8h vchunk = *(const v8h*)&outT[row * 136 + seg * 8];
    *(v8h*)&h1[((size_t)p * NBATCH + rowbase + row) * H1DIM + seg * 8] = vchunk;
  }
  if (tid < H1DIM) {
    atomicAdd(&gSum[p * H1DIM + tid], rSum[tid]);
    atomicAdd(&gSqr[p * H1DIM + tid], rSqr[tid]);
  }
}

// ---------------------------------------------------------------------------
// GEMM2: h2[p,n,512] = relu(a1*h1+c1) @ W7[p]^T + b7 ; + batch stats of h2
// grid (4, N/128, P): 128x128 tile, K=128 in 4 steps of 32, double-buffered LDS
// ---------------------------------------------------------------------------
struct Tiles2 { _Float16 A[2][128 * 40]; _Float16 B[2][128 * 40]; };
union Sh2 { Tiles2 t; _Float16 outT[128 * 136]; };

__global__ void __launch_bounds__(256)
gemm2_kernel(const _Float16* __restrict__ h1, const _Float16* __restrict__ W7h,
             const float* __restrict__ b7,
             const float* __restrict__ cAg, const float* __restrict__ cCg,
             _Float16* __restrict__ h2, float* __restrict__ gSum, float* __restrict__ gSqr)
{
  __shared__ Sh2 sh;
  __shared__ float cA[H1DIM], cC[H1DIM], bias[128], rSum[128], rSqr[128];

  const int tid  = threadIdx.x;
  const int lane = tid & 31, wave = tid >> 5;
  const int l15  = lane & 15, hl = lane >> 4;
  const int colbase = blockIdx.x * 128;
  const int rowbase = blockIdx.y * 128;
  const int p = blockIdx.z;

  if (tid < 128) {
    cA[tid]   = cAg[p * H1DIM + tid];
    cC[tid]   = cCg[p * H1DIM + tid];
    bias[tid] = b7[p * H2DIM + colbase + tid];
    rSum[tid] = 0.0f; rSqr[tid] = 0.0f;
  }
  __syncthreads();

  const int ar  = tid >> 1;          // staging row/col 0..127
  const int akc = (tid & 1) * 16;    // staging K chunk {0,16}
  const _Float16* aSrc = &h1[((size_t)p * NBATCH + rowbase + ar) * H1DIM + akc];
  const _Float16* bSrc = &W7h[((size_t)(p * H2DIM + colbase + ar)) * H1DIM + akc];

  v8f acc[8];
#pragma unroll
  for (int nt = 0; nt < 8; ++nt)
#pragma unroll
    for (int i = 0; i < 8; ++i) acc[nt][i] = 0.0f;

  // prologue: load + stage K-chunk 0 into buffer 0
  v8h a0 = *(const v8h*)(aSrc),     a1 = *(const v8h*)(aSrc + 8);
  v8h b0 = *(const v8h*)(bSrc),     b1 = *(const v8h*)(bSrc + 8);
  {
    v8h y0, y1;
#pragma unroll
    for (int i = 0; i < 8; ++i) {
      float v0 = fmaf((float)a0[i], cA[akc + i],     cC[akc + i]);
      float v1 = fmaf((float)a1[i], cA[akc + 8 + i], cC[akc + 8 + i]);
      y0[i] = (_Float16)(v0 > 0.0f ? v0 : 0.0f);
      y1[i] = (_Float16)(v1 > 0.0f ? v1 : 0.0f);
    }
    *(v8h*)&sh.t.A[0][ar * 40 + akc]     = y0;
    *(v8h*)&sh.t.A[0][ar * 40 + akc + 8] = y1;
    *(v8h*)&sh.t.B[0][ar * 40 + akc]     = b0;
    *(v8h*)&sh.t.B[0][ar * 40 + akc + 8] = b1;
  }
  __syncthreads();

  for (int it = 0; it < 4; ++it) {
    const int cur = it & 1;
    const int kbn = (it + 1) * 32;
    if (it < 3) {            // issue next chunk's global loads before the WMMAs
      a0 = *(const v8h*)(aSrc + kbn); a1 = *(const v8h*)(aSrc + kbn + 8);
      b0 = *(const v8h*)(bSrc + kbn); b1 = *(const v8h*)(bSrc + kbn + 8);
    }
    if (it < 2) {            // prefetch two chunks ahead
      __builtin_prefetch(aSrc + kbn + 32, 0, 1);
      __builtin_prefetch(bSrc + kbn + 32, 0, 1);
    }

    HFrag af;
    af.p[0] = *(const v8h*)&sh.t.A[cur][(wave * 16 + l15) * 40 + hl * 8];
    af.p[1] = *(const v8h*)&sh.t.A[cur][(wave * 16 + l15) * 40 + 16 + hl * 8];
#pragma unroll
    for (int nt = 0; nt < 8; ++nt) {
      HFrag bf;
      bf.p[0] = *(const v8h*)&sh.t.B[cur][(nt * 16 + l15) * 40 + hl * 8];
      bf.p[1] = *(const v8h*)&sh.t.B[cur][(nt * 16 + l15) * 40 + 16 + hl * 8];
      acc[nt] = wmma_f16(af, bf, acc[nt]);
    }

    if (it < 3) {            // stage next chunk into the other buffer
      const int nxt = cur ^ 1;
      v8h y0, y1;
#pragma unroll
      for (int i = 0; i < 8; ++i) {
        float v0 = fmaf((float)a0[i], cA[kbn + akc + i],     cC[kbn + akc + i]);
        float v1 = fmaf((float)a1[i], cA[kbn + akc + 8 + i], cC[kbn + akc + 8 + i]);
        y0[i] = (_Float16)(v0 > 0.0f ? v0 : 0.0f);
        y1[i] = (_Float16)(v1 > 0.0f ? v1 : 0.0f);
      }
      *(v8h*)&sh.t.A[nxt][ar * 40 + akc]     = y0;
      *(v8h*)&sh.t.A[nxt][ar * 40 + akc + 8] = y1;
      *(v8h*)&sh.t.B[nxt][ar * 40 + akc]     = b0;
      *(v8h*)&sh.t.B[nxt][ar * 40 + akc + 8] = b1;
    }
    __syncthreads();         // one barrier per K-step (double-buffered)
  }

  // epilogue: stats + repack into LDS (union reuse is safe: barrier above)
#pragma unroll
  for (int nt = 0; nt < 8; ++nt) {
    const int nl = nt * 16 + l15;
    const float bb = bias[nl];
    float ls = 0.0f, lq = 0.0f;
#pragma unroll
    for (int r = 0; r < 8; ++r) {
      float raw = acc[nt][r] + bb;
      sh.outT[(wave * 16 + hl * 8 + r) * 136 + nl] = (_Float16)raw;
      ls += raw; lq += raw * raw;
    }
    atomicAdd(&rSum[nl], ls);
    atomicAdd(&rSqr[nl], lq);
  }
  __syncthreads();
  // cooperative coalesced store: 128 rows x 16 b128-chunks
#pragma unroll
  for (int i = 0; i < 8; ++i) {
    const int c = tid + i * 256;
    const int row = c >> 4, seg = c & 15;
    v8h vchunk = *(const v8h*)&sh.outT[row * 136 + seg * 8];
    *(v8h*)&h2[((size_t)p * NBATCH + rowbase + row) * H2DIM + colbase + seg * 8] = vchunk;
  }
  if (tid < 128) {
    atomicAdd(&gSum[p * H2DIM + colbase + tid], rSum[tid]);
    atomicAdd(&gSqr[p * H2DIM + colbase + tid], rSqr[tid]);
  }
}

// ---------------------------------------------------------------------------
// GEMM3: out[n,p,c] = sigmoid(relu(a2*h2+c2) @ W8[p]^T + b8)
// grid (2, N/128, P): 128x112 tile (7 col tiles), K=512 in 16 steps, dbl-buffered
// ---------------------------------------------------------------------------
struct Tiles3 { _Float16 A[2][128 * 40]; _Float16 B[2][112 * 40]; };
union Sh3 { Tiles3 t; float outT[128 * 116]; };

__global__ void __launch_bounds__(256)
gemm3_kernel(const _Float16* __restrict__ h2, const _Float16* __restrict__ W8h,
             const float* __restrict__ b8,
             const float* __restrict__ cAg, const float* __restrict__ cCg,
             float* __restrict__ out)
{
  __shared__ Sh3 sh;
  __shared__ float cA[H2DIM], cC[H2DIM], bias[112];

  const int tid  = threadIdx.x;
  const int lane = tid & 31, wave = tid >> 5;
  const int l15  = lane & 15, hl = lane >> 4;
  const int colbase = blockIdx.x * 112;
  const int rowbase = blockIdx.y * 128;
  const int p = blockIdx.z;

  for (int i = tid; i < H2DIM; i += 256) {
    cA[i] = cAg[p * H2DIM + i];
    cC[i] = cCg[p * H2DIM + i];
  }
  if (tid < 112) bias[tid] = b8[p * CDIM + colbase + tid];
  __syncthreads();

  const int ar  = tid >> 1;
  const int akc = (tid & 1) * 16;
  const _Float16* aSrc = &h2[((size_t)p * NBATCH + rowbase + ar) * H2DIM + akc];
  // B staging: 448 chunks of 8 halves; thread handles chunk tid and (tid+256 if tid<192)
  const int n0 = tid >> 2,          kc0 = (tid & 3) * 8;
  const int n1 = (tid + 256) >> 2,  kc1 = ((tid + 256) & 3) * 8;
  const bool hasB1 = (tid < 192);
  const _Float16* bSrc0 = &W8h[((size_t)(p * CDIM + colbase + n0)) * H2DIM + kc0];
  const _Float16* bSrc1 = &W8h[((size_t)(p * CDIM + colbase + (hasB1 ? n1 : n0))) * H2DIM + kc1];

  v8f acc[7];
#pragma unroll
  for (int nt = 0; nt < 7; ++nt)
#pragma unroll
    for (int i = 0; i < 8; ++i) acc[nt][i] = 0.0f;

  // prologue: load + stage K-chunk 0 into buffer 0
  v8h a0 = *(const v8h*)(aSrc), a1 = *(const v8h*)(aSrc + 8);
  v8h bR0 = *(const v8h*)(bSrc0), bR1 = *(const v8h*)(bSrc1);
  {
    v8h y0, y1;
#pragma unroll
    for (int i = 0; i < 8; ++i) {
      float v0 = fmaf((float)a0[i], cA[akc + i],     cC[akc + i]);
      float v1 = fmaf((float)a1[i], cA[akc + 8 + i], cC[akc + 8 + i]);
      y0[i] = (_Float16)(v0 > 0.0f ? v0 : 0.0f);
      y1[i] = (_Float16)(v1 > 0.0f ? v1 : 0.0f);
    }
    *(v8h*)&sh.t.A[0][ar * 40 + akc]     = y0;
    *(v8h*)&sh.t.A[0][ar * 40 + akc + 8] = y1;
    *(v8h*)&sh.t.B[0][n0 * 40 + kc0] = bR0;
    if (hasB1) *(v8h*)&sh.t.B[0][n1 * 40 + kc1] = bR1;
  }
  __syncthreads();

  for (int it = 0; it < 16; ++it) {
    const int cur = it & 1;
    const int kbn = (it + 1) * 32;
    if (it < 15) {
      a0  = *(const v8h*)(aSrc + kbn);  a1  = *(const v8h*)(aSrc + kbn + 8);
      bR0 = *(const v8h*)(bSrc0 + kbn); bR1 = *(const v8h*)(bSrc1 + kbn);
    }
    if (it < 14) {
      __builtin_prefetch(aSrc + kbn + 32, 0, 1);
      __builtin_prefetch(bSrc0 + kbn + 32, 0, 1);
    }

    HFrag af;
    af.p[0] = *(const v8h*)&sh.t.A[cur][(wave * 16 + l15) * 40 + hl * 8];
    af.p[1] = *(const v8h*)&sh.t.A[cur][(wave * 16 + l15) * 40 + 16 + hl * 8];
#pragma unroll
    for (int nt = 0; nt < 7; ++nt) {
      HFrag bf;
      bf.p[0] = *(const v8h*)&sh.t.B[cur][(nt * 16 + l15) * 40 + hl * 8];
      bf.p[1] = *(const v8h*)&sh.t.B[cur][(nt * 16 + l15) * 40 + 16 + hl * 8];
      acc[nt] = wmma_f16(af, bf, acc[nt]);
    }

    if (it < 15) {
      const int nxt = cur ^ 1;
      v8h y0, y1;
#pragma unroll
      for (int i = 0; i < 8; ++i) {
        float v0 = fmaf((float)a0[i], cA[kbn + akc + i],     cC[kbn + akc + i]);
        float v1 = fmaf((float)a1[i], cA[kbn + akc + 8 + i], cC[kbn + akc + 8 + i]);
        y0[i] = (_Float16)(v0 > 0.0f ? v0 : 0.0f);
        y1[i] = (_Float16)(v1 > 0.0f ? v1 : 0.0f);
      }
      *(v8h*)&sh.t.A[nxt][ar * 40 + akc]     = y0;
      *(v8h*)&sh.t.A[nxt][ar * 40 + akc + 8] = y1;
      *(v8h*)&sh.t.B[nxt][n0 * 40 + kc0] = bR0;
      if (hasB1) *(v8h*)&sh.t.B[nxt][n1 * 40 + kc1] = bR1;
    }
    __syncthreads();
  }

  // epilogue: bias + sigmoid, repack f32 tile into LDS, coalesced b128 stores
#pragma unroll
  for (int nt = 0; nt < 7; ++nt) {
    const int cl = nt * 16 + l15;
    const float bb = bias[cl];
#pragma unroll
    for (int r = 0; r < 8; ++r) {
      float raw = acc[nt][r] + bb;
      sh.outT[(wave * 16 + hl * 8 + r) * 116 + cl] = 1.0f / (1.0f + __expf(-raw));
    }
  }
  __syncthreads();
  // 128 rows x 28 v4f-chunks = 3584 chunks, 14 per thread
#pragma unroll
  for (int i = 0; i < 14; ++i) {
    const int c = tid + i * 256;
    const int row = c / 28;
    const int seg = c - row * 28;
    v4f vchunk = *(const v4f*)&sh.outT[row * 116 + seg * 4];
    *(v4f*)&out[(size_t)(rowbase + row) * (PBLK * CDIM) + p * CDIM + colbase + seg * 4] = vchunk;
  }
}

// ---------------------------------------------------------------------------
// Host side
// ---------------------------------------------------------------------------
// Workspace layout (bytes, all offsets multiple of 256)
static const size_t OFF_ZH  = 0;                       // N*16 f16        = 1,048,576
static const size_t OFF_W6H = 1048576;                 // 8*128*16 f16    =    32,768
static const size_t OFF_W7H = 1081344;                 // 8*512*128 f16   = 1,048,576
static const size_t OFF_W8H = 2129920;                 // 8*224*512 f16   = 1,835,008
static const size_t OFF_S1  = 3964928;                 // sum+sqr P*128 f32 (8,192)
static const size_t OFF_S2  = 3973120;                 // sum+sqr P*512 f32 (32,768)
static const size_t OFF_C1  = 4005888;                 // a+c P*128 f32 (8,192)
static const size_t OFF_C2  = 4014080;                 // a+c P*512 f32 (32,768)
static const size_t OFF_H1  = 4046848;                 // P*N*128 f16     = 67,108,864
static const size_t OFF_H2  = 71155712;                // P*N*512 f16     = 268,435,456

extern "C" void kernel_launch(void* const* d_in, const int* in_sizes, int n_in,
                              void* d_out, int out_size, void* d_ws, size_t ws_size,
                              hipStream_t stream) {
  const float* z   = (const float*)d_in[0];
  const float* W6  = (const float*)d_in[1];
  const float* b6  = (const float*)d_in[2];
  const float* g6  = (const float*)d_in[3];
  const float* be6 = (const float*)d_in[4];
  const float* W7  = (const float*)d_in[5];
  const float* b7  = (const float*)d_in[6];
  const float* g7  = (const float*)d_in[7];
  const float* be7 = (const float*)d_in[8];
  const float* W8  = (const float*)d_in[9];
  const float* b8  = (const float*)d_in[10];
  float* out = (float*)d_out;

  char* ws = (char*)d_ws;
  _Float16* zh   = (_Float16*)(ws + OFF_ZH);
  _Float16* W6h  = (_Float16*)(ws + OFF_W6H);
  _Float16* W7h  = (_Float16*)(ws + OFF_W7H);
  _Float16* W8h  = (_Float16*)(ws + OFF_W8H);
  float* s1sum = (float*)(ws + OFF_S1);
  float* s1sqr = s1sum + PBLK * H1DIM;
  float* s2sum = (float*)(ws + OFF_S2);
  float* s2sqr = s2sum + PBLK * H2DIM;
  float* c1a = (float*)(ws + OFF_C1);
  float* c1c = c1a + PBLK * H1DIM;
  float* c2a = (float*)(ws + OFF_C2);
  float* c2c = c2a + PBLK * H2DIM;
  _Float16* h1 = (_Float16*)(ws + OFF_H1);
  _Float16* h2 = (_Float16*)(ws + OFF_H2);

  const float invN = 1.0f / (float)NBATCH;

  // 1) convert inputs/weights to f16; zero stats accumulators
  {
    int n;
    n = NBATCH * ZDIM;
    cvt_f16_kernel<<<(n + 255) / 256, 256, 0, stream>>>(z, zh, n);
    n = PBLK * H1DIM * ZDIM;
    cvt_f16_kernel<<<(n + 255) / 256, 256, 0, stream>>>(W6, W6h, n);
    n = PBLK * H2DIM * H1DIM;
    cvt_f16_kernel<<<(n + 255) / 256, 256, 0, stream>>>(W7, W7h, n);
    n = PBLK * CDIM * H2DIM;
    cvt_f16_kernel<<<(n + 255) / 256, 256, 0, stream>>>(W8, W8h, n);
    n = 2 * PBLK * H1DIM + 2 * PBLK * H2DIM;  // contiguous S1 then S2 region
    zero_f32_kernel<<<(n + 255) / 256, 256, 0, stream>>>(s1sum, n);
  }

  // 2) layer 1 GEMM + stats
  gemm1_kernel<<<dim3(NBATCH / 128, PBLK), 256, 0, stream>>>(zh, W6h, b6, h1, s1sum, s1sqr);
  bn_finalize_kernel<<<(PBLK * H1DIM + 255) / 256, 256, 0, stream>>>(
      s1sum, s1sqr, g6, be6, c1a, c1c, PBLK * H1DIM, invN);

  // 3) layer 2 GEMM (fused BN1+ReLU on load) + stats
  gemm2_kernel<<<dim3(H2DIM / 128, NBATCH / 128, PBLK), 256, 0, stream>>>(
      h1, W7h, b7, c1a, c1c, h2, s2sum, s2sqr);
  bn_finalize_kernel<<<(PBLK * H2DIM + 255) / 256, 256, 0, stream>>>(
      s2sum, s2sqr, g7, be7, c2a, c2c, PBLK * H2DIM, invN);

  // 4) layer 3 GEMM (fused BN2+ReLU on load) + bias + sigmoid + transpose store
  gemm3_kernel<<<dim3(CDIM / 112, NBATCH / 128, PBLK), 256, 0, stream>>>(
      h2, W8h, b8, c2a, c2c, out);
}
